// Gemma3Attention_48412871360999
// MI455X (gfx1250) — compile-verified, matrix operand order checked by reading, not run on hardware
//
#include <hip/hip_runtime.h>
#include <hip/hip_bf16.h>

// ---------------- constants (match reference) ----------------
#define BATCH   2
#define LSEQ    2048
#define HID     2560
#define NH      8
#define NKV     4
#define HD      256
#define WIN     512
#define QK_SCALE 0.0625f      // 256^-0.5
#define SOFTCAP  50.0f

typedef __attribute__((ext_vector_type(16))) __bf16        bf16x16;
typedef __attribute__((ext_vector_type(8)))  float         f32x8;
typedef __attribute__((ext_vector_type(4)))  unsigned int  u32x4;

union FragU { bf16x16 v; u32x4 u[2]; };

static __device__ __forceinline__ unsigned short f2bf(float f) {
    unsigned int u = __float_as_uint(f);
    u += 0x7FFFu + ((u >> 16) & 1u);      // round-to-nearest-even
    return (unsigned short)(u >> 16);
}

// gfx1250 async global->LDS copy (ASYNCcnt-tracked); per-lane 16B transfer.
static __device__ __forceinline__ void async_ld16(unsigned lds_off, const unsigned short* g) {
    asm volatile("global_load_async_to_lds_b128 %0, %1, off"
                 :: "v"(lds_off), "v"(g) : "memory");
}
static __device__ __forceinline__ void wait_async0() {
    asm volatile("s_wait_asynccnt 0x0" ::: "memory");
}

// ---------------- fp32 -> bf16 elementwise ----------------
__global__ __launch_bounds__(256) void f2bf_kernel(const float* __restrict__ s,
                                                   unsigned short* __restrict__ d, int n) {
    int i = blockIdx.x * 256 + threadIdx.x;
    if (i < n) d[i] = f2bf(s[i]);
}

// ---------------- tiled bf16 WMMA GEMM:  C(M,N) = A(M,K) * B(N,K)^T ----------------
// A row-major (M,K), B row-major (N,K) (weight layout), C fp32 row-major (M,N).
// Block: 256 threads = 8 waves; block tile 128x256x32; wave computes 64x64 (4x4 wmma tiles).
// LDS tiles are double-buffered and filled with global_load_async_to_lds_b128.
__global__ __launch_bounds__(256) void gemm_bf16_kernel(
    const unsigned short* __restrict__ A, const unsigned short* __restrict__ B,
    float* __restrict__ C, int M, int N, int K)
{
    __shared__ __align__(16) unsigned char As[2][128 * 64];   // 128 rows x 32 bf16, xor-swizzled 16B chunks
    __shared__ __align__(16) unsigned char Bs[2][256 * 64];   // 256 rows x 32 bf16

    const int nb = N >> 8;
    const int bx = blockIdx.x % nb;
    const int by = blockIdx.x / nb;
    const int tid  = threadIdx.x;
    const int wv   = tid >> 5;
    const int lane = tid & 31;
    const int half = lane >> 4;
    const int mn   = lane & 15;
    const int wm = wv >> 2;          // 0..1  -> 64-row slab
    const int wn = wv & 3;           // 0..3  -> 64-col slab
    const int rowA0 = by << 7;
    const int colB0 = bx << 8;

    const unsigned short* Ag = A + (size_t)rowA0 * K;
    const unsigned short* Bg = B + (size_t)colB0 * K;
    const unsigned aL0 = (unsigned)(size_t)&As[0][0];
    const unsigned aL1 = (unsigned)(size_t)&As[1][0];
    const unsigned bL0 = (unsigned)(size_t)&Bs[0][0];
    const unsigned bL1 = (unsigned)(size_t)&Bs[1][0];

    f32x8 acc[4][4] = {};

    // stage one 32-wide k-slice of A (128 rows) and B (256 rows) via async copy
    auto stage = [&](int kk, unsigned aL, unsigned bL) {
#pragma unroll
        for (int j = 0; j < 2; j++) {          // A: 512 16B chunks / 256 threads
            int c = tid + j * 256;
            int row = c >> 2, cc = c & 3;
            async_ld16(aL + row * 64 + ((cc ^ (row & 3)) << 4),
                       Ag + (size_t)row * K + kk + cc * 8);
        }
#pragma unroll
        for (int j = 0; j < 4; j++) {          // B: 1024 16B chunks / 256 threads
            int c = tid + j * 256;
            int row = c >> 2, cc = c & 3;
            async_ld16(bL + row * 64 + ((cc ^ (row & 3)) << 4),
                       Bg + (size_t)row * K + kk + cc * 8);
        }
    };

    const int nk = K >> 5;
    stage(0, aL0, bL0);

    for (int i = 0; i < nk; i++) {
        wait_async0();                // this wave's async writes for tile i are in LDS
        __syncthreads();              // all waves' tile-i data visible; tile i-1 reads drained
        if (i + 1 < nk)
            stage((i + 1) << 5, ((i + 1) & 1) ? aL1 : aL0, ((i + 1) & 1) ? bL1 : bL0);

        const unsigned char* Ab = As[i & 1];
        const unsigned char* Bb = Bs[i & 1];
        FragU afr[4], bfr[4];
#pragma unroll
        for (int tm = 0; tm < 4; tm++) {
            int r = wm * 64 + tm * 16 + mn; int sw = r & 3;
            afr[tm].u[0] = *(const u32x4*)(Ab + r * 64 + ((half ^ sw) << 4));
            afr[tm].u[1] = *(const u32x4*)(Ab + r * 64 + (((2 + half) ^ sw) << 4));
        }
#pragma unroll
        for (int tn = 0; tn < 4; tn++) {
            int r = wn * 64 + tn * 16 + mn; int sw = r & 3;
            bfr[tn].u[0] = *(const u32x4*)(Bb + r * 64 + ((half ^ sw) << 4));
            bfr[tn].u[1] = *(const u32x4*)(Bb + r * 64 + (((2 + half) ^ sw) << 4));
        }
#pragma unroll
        for (int tm = 0; tm < 4; tm++)
#pragma unroll
            for (int tn = 0; tn < 4; tn++)
                acc[tm][tn] = __builtin_amdgcn_wmma_f32_16x16x32_bf16(
                    false, afr[tm].v, false, bfr[tn].v, (short)0, acc[tm][tn], false, false);
    }

#pragma unroll
    for (int tm = 0; tm < 4; tm++)
#pragma unroll
        for (int tn = 0; tn < 4; tn++) {
            int n = colB0 + wn * 64 + tn * 16 + mn;
            int m0 = rowA0 + wm * 64 + tm * 16 + 8 * half;
#pragma unroll
            for (int r = 0; r < 8; r++)
                C[(size_t)(m0 + r) * N + n] = acc[tm][tn][r];
        }
}

// ---------------- fused RMSNorm(1+w) + RoPE; fp32 (B,L,H*D) -> bf16 (B,H,L,D) ----------------
__global__ __launch_bounds__(256) void norm_rope_kernel(const float* __restrict__ src,
    unsigned short* __restrict__ dst, const float* __restrict__ nw, int H)
{
    int vec  = blockIdx.x * 8 + (threadIdx.x >> 5);   // one wave per 256-elem head vector
    int lane = threadIdx.x & 31;
    int h = vec % H;
    int l = (vec / H) % LSEQ;
    int b = vec / (H * LSEQ);
    const float* xp = src + ((size_t)(b * LSEQ + l) * H + h) * HD;
    int d0 = lane * 8;

    float x[8]; float ss = 0.f;
#pragma unroll
    for (int j = 0; j < 8; j++) { x[j] = xp[d0 + j]; ss += x[j] * x[j]; }
#pragma unroll
    for (int m = 16; m >= 1; m >>= 1) ss += __shfl_xor(ss, m, 32);
    float inv = rsqrtf(ss * (1.f / 256.f) + 1e-6f);

    float nx[8];
#pragma unroll
    for (int j = 0; j < 8; j++) nx[j] = x[j] * inv * (1.f + nw[d0 + j]);

    unsigned short* op = dst + (((size_t)(b * H + h) * LSEQ + l) * HD);
#pragma unroll
    for (int j = 0; j < 8; j++) {
        float other = __shfl_xor(nx[j], 16, 32);      // partner at d ^ 128
        int d = d0 + j;
        float rot = (d < 128) ? -other : other;
        float ang = (float)l * powf(10000.f, -(float)(d & 127) * (1.f / 128.f));
        float o = nx[j] * cosf(ang) + rot * sinf(ang);
        op[d] = f2bf(o);
    }
}

// ---------------- V: fp32 (B,L,Hkv*D) -> bf16 transposed (B,Hkv,D,L) ----------------
__global__ __launch_bounds__(256) void vtrans_kernel(const float* __restrict__ src,
                                                     unsigned short* __restrict__ dst) {
    size_t i = (size_t)blockIdx.x * 256 + threadIdx.x;   // B*L*NKV*D elements
    int d = (int)(i % HD);
    int h = (int)((i / HD) % NKV);
    int l = (int)((i / (HD * NKV)) % LSEQ);
    int b = (int)(i / ((size_t)HD * NKV * LSEQ));
    dst[(((size_t)(b * NKV + h) * HD + d) * LSEQ) + l] = f2bf(src[i]);
}

// ---------------- attention: one workgroup (4 waves) per (b, h, 16-query tile) ----------------
__global__ __launch_bounds__(128) void attn_kernel(
    const unsigned short* __restrict__ qm, const unsigned short* __restrict__ km,
    const unsigned short* __restrict__ vt, const float* __restrict__ mask,
    unsigned short* __restrict__ outp)
{
    __shared__ __align__(16) float          Ssm[16][WIN + 4];
    __shared__ __align__(16) unsigned short Psm[16][WIN + 16];

    const int bid = blockIdx.x;
    const int qt = bid & 127;
    const int h  = (bid >> 7) & 7;
    const int b  = bid >> 10;
    const int hk = h >> 1;
    const int wv = threadIdx.x >> 5, lane = threadIdx.x & 31;
    const int half = lane >> 4, mn = lane & 15;

    const unsigned short* qp = qm + (((size_t)(b * NH + h) * LSEQ) + qt * 16) * HD;
    const unsigned short* kp = km + (((size_t)(b * NKV + hk) * LSEQ) + (LSEQ - WIN)) * HD;
    const unsigned short* vp = vt + ((size_t)(b * NKV + hk) * HD) * LSEQ + (LSEQ - WIN);

    // Q fragments in registers: 8 k-steps of 32 over D=256
    FragU qf[8];
#pragma unroll
    for (int ds = 0; ds < 8; ds++) {
        const unsigned short* base = qp + (size_t)mn * HD + ds * 32 + half * 8;
        qf[ds].u[0] = *(const u32x4*)base;
        qf[ds].u[1] = *(const u32x4*)(base + 16);
    }

    // S = softcap(QK^T * scale) + mask, window of 512 keys, 32 key tiles split over 4 waves
    for (int kt = wv; kt < 32; kt += 4) {
        f32x8 acc = {};
        const unsigned short* kbase = kp + (size_t)(kt * 16 + mn) * HD;
#pragma unroll
        for (int ds = 0; ds < 8; ds++) {
            FragU kf;
            const unsigned short* kb = kbase + ds * 32 + half * 8;
            kf.u[0] = *(const u32x4*)kb;
            kf.u[1] = *(const u32x4*)(kb + 16);
            acc = __builtin_amdgcn_wmma_f32_16x16x32_bf16(
                false, qf[ds].v, false, kf.v, (short)0, acc, false, false);
        }
        int col = kt * 16 + mn;
        int gq  = qt * 16;
#pragma unroll
        for (int r = 0; r < 8; r++) {
            int row = r + 8 * half;
            float s = SOFTCAP * tanhf(acc[r] * QK_SCALE * (1.0f / SOFTCAP));
            s += mask[(size_t)(gq + row) * LSEQ + (LSEQ - WIN) + col];
            Ssm[row][col] = s;
        }
    }
    __syncthreads();

    // softmax over 512 in fp32; 8 lanes per row, 64 cols each
    {
        int row = wv * 4 + (lane >> 3);
        int c0  = (lane & 7) * 64;
        float mx = -3.0e38f;
        for (int c = c0; c < c0 + 64; c++) mx = fmaxf(mx, Ssm[row][c]);
#pragma unroll
        for (int m = 4; m >= 1; m >>= 1) mx = fmaxf(mx, __shfl_xor(mx, m, 32));
        float sum = 0.f;
        for (int c = c0; c < c0 + 64; c++) { float e = __expf(Ssm[row][c] - mx); Ssm[row][c] = e; sum += e; }
#pragma unroll
        for (int m = 4; m >= 1; m >>= 1) sum += __shfl_xor(sum, m, 32);
        float is = 1.f / sum;
        for (int c = c0; c < c0 + 64; c++) Psm[row][c] = f2bf(Ssm[row][c] * is);
    }
    __syncthreads();

    // O = P(16x512) @ V(512x256); 16 n-tiles split over 4 waves; K=512 -> 16 wmma steps
    for (int nt = wv; nt < 16; nt += 4) {
        f32x8 acc = {};
        const unsigned short* vrow = vp + (size_t)(nt * 16 + mn) * LSEQ;
#pragma unroll
        for (int ks = 0; ks < 16; ks++) {
            FragU pf, vf;
            const unsigned short* pb = &Psm[mn][ks * 32 + half * 8];
            pf.u[0] = *(const u32x4*)pb;
            pf.u[1] = *(const u32x4*)(pb + 16);
            const unsigned short* vb = vrow + ks * 32 + half * 8;
            vf.u[0] = *(const u32x4*)vb;
            vf.u[1] = *(const u32x4*)(vb + 16);
            acc = __builtin_amdgcn_wmma_f32_16x16x32_bf16(
                false, pf.v, false, vf.v, (short)0, acc, false, false);
        }
        int n = nt * 16 + mn;
#pragma unroll
        for (int r = 0; r < 8; r++) {
            int row = r + 8 * half;
            outp[((size_t)(b * LSEQ + qt * 16 + row)) * (NH * HD) + h * HD + n] = f2bf(acc[r]);
        }
    }
}

// ---------------- host launcher ----------------
extern "C" void kernel_launch(void* const* d_in, const int* in_sizes, int n_in,
                              void* d_out, int out_size, void* d_ws, size_t ws_size,
                              hipStream_t stream) {
    (void)in_sizes; (void)n_in; (void)out_size; (void)ws_size;
    const float* x    = (const float*)d_in[0];
    const float* mask = (const float*)d_in[1];
    const float* q_w  = (const float*)d_in[2];
    const float* k_w  = (const float*)d_in[3];
    const float* v_w  = (const float*)d_in[4];
    const float* o_w  = (const float*)d_in[5];
    const float* qnw  = (const float*)d_in[6];
    const float* knw  = (const float*)d_in[7];
    float* outv = (float*)d_out;

    char* ws = (char*)d_ws;
    size_t off = 0;
    auto alloc = [&](size_t bytes) { size_t o = off; off += (bytes + 255) & ~(size_t)255; return o; };

    const int M  = BATCH * LSEQ;             // 4096
    const int NQ = NH * HD;                  // 2048
    const int NK = NKV * HD;                 // 1024

    unsigned short* xb   = (unsigned short*)(ws + alloc((size_t)M * HID * 2));
    unsigned short* qwb  = (unsigned short*)(ws + alloc((size_t)NQ * HID * 2));
    unsigned short* kwb  = (unsigned short*)(ws + alloc((size_t)NK * HID * 2));
    unsigned short* vwb  = (unsigned short*)(ws + alloc((size_t)NK * HID * 2));
    unsigned short* owb  = (unsigned short*)(ws + alloc((size_t)HID * NQ * 2));
    float*          qpr  = (float*)(ws + alloc((size_t)M * NQ * 4));
    float*          kpr  = (float*)(ws + alloc((size_t)M * NK * 4));
    float*          vpr  = (float*)(ws + alloc((size_t)M * NK * 4));
    unsigned short* qbb  = (unsigned short*)(ws + alloc((size_t)M * NQ * 2));
    unsigned short* kbb  = (unsigned short*)(ws + alloc((size_t)M * NK * 2));
    unsigned short* vtb  = (unsigned short*)(ws + alloc((size_t)M * NK * 2));
    unsigned short* aout = (unsigned short*)(ws + alloc((size_t)M * NQ * 2));

    auto cvt = [&](const float* s, unsigned short* d, int n) {
        f2bf_kernel<<<(n + 255) / 256, 256, 0, stream>>>(s, d, n);
    };
    cvt(x,   xb,  M * HID);
    cvt(q_w, qwb, NQ * HID);
    cvt(k_w, kwb, NK * HID);
    cvt(v_w, vwb, NK * HID);
    cvt(o_w, owb, HID * NQ);

    gemm_bf16_kernel<<<(M / 128) * (NQ / 256), 256, 0, stream>>>(xb, qwb, qpr, M, NQ, HID);
    gemm_bf16_kernel<<<(M / 128) * (NK / 256), 256, 0, stream>>>(xb, kwb, kpr, M, NK, HID);
    gemm_bf16_kernel<<<(M / 128) * (NK / 256), 256, 0, stream>>>(xb, vwb, vpr, M, NK, HID);

    norm_rope_kernel<<<(BATCH * LSEQ * NH) / 8,  256, 0, stream>>>(qpr, qbb, qnw, NH);
    norm_rope_kernel<<<(BATCH * LSEQ * NKV) / 8, 256, 0, stream>>>(kpr, kbb, knw, NKV);
    vtrans_kernel<<<(M * NK) / 256, 256, 0, stream>>>(vpr, vtb);

    attn_kernel<<<BATCH * NH * (LSEQ / 16), 128, 0, stream>>>(qbb, kbb, vtb, mask, aout);

    gemm_bf16_kernel<<<(M / 128) * (HID / 256), 256, 0, stream>>>(aout, owb, outv, M, HID, NQ);
}
